// SpectralMixer_89240830476289
// MI455X (gfx1250) — matrix operand order repaired and checked by reading, not compiled, not run on hardware
//
#include <hip/hip_runtime.h>
#include <hip/hip_bf16.h>

// Problem constants (match reference)
#define BB 4
#define NN 4096
#define DD 512
#define NUM_STAGES 12

typedef __attribute__((ext_vector_type(16))) __bf16 v16bf;
typedef __attribute__((ext_vector_type(8)))  float  v8f;

union BFFrag { v16bf v; unsigned short u[16]; };

__device__ __forceinline__ unsigned short f2bf(float f) {
    // round-to-nearest-even fp32 -> bf16 (data is finite; NaN not special-cased)
    unsigned int u = __float_as_uint(f);
    u += 0x7FFFu + ((u >> 16) & 1u);
    return (unsigned short)(u >> 16);
}
__device__ __forceinline__ float bf2f(unsigned short h) {
    return __uint_as_float(((unsigned int)h) << 16);
}

// ---------------------------------------------------------------------------
// One-shot: split W_out (512x512 f32, row-major e,k) into bf16 hi/lo planes.
// ---------------------------------------------------------------------------
__global__ __launch_bounds__(256)
void wsplit_kernel(const float* __restrict__ W,
                   unsigned short* __restrict__ wh,
                   unsigned short* __restrict__ wl) {
    const int idx = blockIdx.x * 256 + threadIdx.x;   // grid covers DD*DD exactly
    const float v = W[idx];
    const unsigned short h = f2bf(v);
    wh[idx] = h;
    wl[idx] = f2bf(v - bf2f(h));
}

// ---------------------------------------------------------------------------
// Butterfly pass 1: stages 0..5, contiguous 64-groups, f32 -> f32.
// 64(position) x 64(channel) LDS tile; 6 stages resident.
// ---------------------------------------------------------------------------
__global__ __launch_bounds__(256)
void butterfly_p1_kernel(const float* __restrict__ src, float* __restrict__ dst,
                         const float* __restrict__ sw) {
    __shared__ float tile[64][64];

    const int tid = threadIdx.x;
    const int c  = tid & 63;
    const int pg = tid >> 6;

    const int g  = blockIdx.x;
    const int d0 = blockIdx.y * 64;
    const int b  = blockIdx.z;

    const int base = g * 64;
    const size_t planeB = (size_t)b * NN * DD;

    #pragma unroll
    for (int i = 0; i < 16; ++i) {
        const int p = pg * 16 + i;
        tile[p][c] = src[planeB + (size_t)(base + p) * DD + d0 + c];
    }

    for (int s = 0; s < 6; ++s) {
        __syncthreads();
        #pragma unroll
        for (int i = 0; i < 8; ++i) {
            const int q = pg * 8 + i;
            const int lmask = (1 << s) - 1;
            const int p0 = ((q >> s) << (s + 1)) | (q & lmask);
            const int p1 = p0 | (1 << s);
            const int n0 = base + p0;
            const int r = ((n0 >> (s + 1)) << s) | (n0 & ((1 << s) - 1));
            const float4 wv = *(const float4*)(sw + (size_t)s * (NN / 2) * 4
                                                  + (size_t)r * 4);
            const float a  = tile[p0][c];
            const float bb = tile[p1][c];
            tile[p0][c] = wv.x * a + wv.y * bb;
            tile[p1][c] = wv.z * a + wv.w * bb;
        }
    }
    __syncthreads();

    #pragma unroll
    for (int i = 0; i < 16; ++i) {
        const int p = pg * 16 + i;
        dst[planeB + (size_t)(base + p) * DD + d0 + c] = tile[p][c];
    }
}

// ---------------------------------------------------------------------------
// Butterfly pass 2: stages 6..11, stride-64 groups, f32 -> bf16 hi/lo planes.
// The split is done here ONCE per element, so the GEMM does zero conversions.
// ---------------------------------------------------------------------------
__global__ __launch_bounds__(256)
void butterfly_p2_kernel(const float* __restrict__ src,
                         unsigned short* __restrict__ yh,
                         unsigned short* __restrict__ yl,
                         const float* __restrict__ sw) {
    __shared__ float tile[64][64];

    const int tid = threadIdx.x;
    const int c  = tid & 63;
    const int pg = tid >> 6;

    const int g  = blockIdx.x;           // low 6 bits of n
    const int d0 = blockIdx.y * 64;
    const int b  = blockIdx.z;

    const size_t planeB = (size_t)b * NN * DD;

    #pragma unroll
    for (int i = 0; i < 16; ++i) {
        const int p = pg * 16 + i;
        const int n = g + p * 64;
        tile[p][c] = src[planeB + (size_t)n * DD + d0 + c];
    }

    for (int sl = 0; sl < 6; ++sl) {
        const int s = 6 + sl;
        __syncthreads();
        #pragma unroll
        for (int i = 0; i < 8; ++i) {
            const int q = pg * 8 + i;
            const int lmask = (1 << sl) - 1;
            const int p0 = ((q >> sl) << (sl + 1)) | (q & lmask);
            const int p1 = p0 | (1 << sl);
            const int n0 = g + p0 * 64;
            const int r = ((n0 >> (s + 1)) << s) | (n0 & ((1 << s) - 1));
            const float4 wv = *(const float4*)(sw + (size_t)s * (NN / 2) * 4
                                                  + (size_t)r * 4);
            const float a  = tile[p0][c];
            const float bb = tile[p1][c];
            tile[p0][c] = wv.x * a + wv.y * bb;
            tile[p1][c] = wv.z * a + wv.w * bb;
        }
    }
    __syncthreads();

    #pragma unroll
    for (int i = 0; i < 16; ++i) {
        const int p = pg * 16 + i;
        const int n = g + p * 64;
        const size_t off = planeB + (size_t)n * DD + d0 + c;
        const float v = tile[p][c];
        const unsigned short h = f2bf(v);
        yh[off] = h;
        yl[off] = f2bf(v - bf2f(h));
    }
}

// ---------------------------------------------------------------------------
// GEMM: out[m][e] = sum_k Y[m][k]*W[e][k] + bias[e];  M=16384, E=K=512.
// bf16x3 emulated fp32: acc += Ah*Bh + Ah*Bl + Al*Bh  (fp32 accumulate).
// Fragments load directly from pre-split bf16 planes (pure b128 loads).
// Wave tile = 16(M) x 64(E); 8 waves stacked in M -> 128 x 64 per workgroup.
// ---------------------------------------------------------------------------
__global__ __launch_bounds__(256)
void gemm_kernel(const unsigned short* __restrict__ yh,
                 const unsigned short* __restrict__ yl,
                 const unsigned short* __restrict__ wh,
                 const unsigned short* __restrict__ wl,
                 const float* __restrict__ bias, float* __restrict__ out) {
    const int K = DD;

    const int lane = threadIdx.x & 31;
    const int wave = threadIdx.x >> 5;
    const int m0   = blockIdx.x * 128 + wave * 16;
    const int e0   = blockIdx.y * 64;

    const int hsel = (lane < 16) ? 0 : 1;
    const int lrow = lane & 15;

    v8f acc[4];
    #pragma unroll
    for (int t = 0; t < 4; ++t) acc[t] = (v8f){};

    const int mrow  = m0 + lrow;
    const int koffA = hsel * 8;     // A lane half: K {0..7,16..23} or {8..15,24..31}
    const int koffB = hsel * 8;     // B lane half: K {0..7,16..23} or {8..15,24..31}

    for (int k0 = 0; k0 < K; k0 += 32) {
        // ---- A fragments: two 16B runs per plane (pre-packed bf16) ----
        const unsigned short* ypH = yh + (size_t)mrow * K + k0 + koffA;
        const unsigned short* ypL = yl + (size_t)mrow * K + k0 + koffA;
        if (k0 + 32 < K) __builtin_prefetch(ypH + 32, 0, 1);  // global_prefetch_b8

        BFFrag ah, al;
        *(uint4*)&ah.u[0] = *(const uint4*)(ypH);
        *(uint4*)&ah.u[8] = *(const uint4*)(ypH + 16);
        *(uint4*)&al.u[0] = *(const uint4*)(ypL);
        *(uint4*)&al.u[8] = *(const uint4*)(ypL + 16);

        // ---- 4 E-tiles: B fragments = two 16B runs per plane ----
        #pragma unroll
        for (int t = 0; t < 4; ++t) {
            const int col = e0 + t * 16 + lrow;
            const unsigned short* wpH = wh + (size_t)col * K + k0 + koffB;
            const unsigned short* wpL = wl + (size_t)col * K + k0 + koffB;

            BFFrag bh, bl;
            *(uint4*)&bh.u[0] = *(const uint4*)(wpH);
            *(uint4*)&bh.u[8] = *(const uint4*)(wpH + 16);
            *(uint4*)&bl.u[0] = *(const uint4*)(wpL);
            *(uint4*)&bl.u[8] = *(const uint4*)(wpL + 16);

            acc[t] = __builtin_amdgcn_wmma_f32_16x16x32_bf16(
                false, ah.v, false, bh.v, (short)0, acc[t], false, false);
            acc[t] = __builtin_amdgcn_wmma_f32_16x16x32_bf16(
                false, ah.v, false, bl.v, (short)0, acc[t], false, false);
            acc[t] = __builtin_amdgcn_wmma_f32_16x16x32_bf16(
                false, al.v, false, bh.v, (short)0, acc[t], false, false);
        }
    }

    // ---- store: C/D layout -> VGPR r: M = r (+8 for lanes 16..31), N = lane&15
    #pragma unroll
    for (int t = 0; t < 4; ++t) {
        const int ncol = e0 + t * 16 + lrow;
        const float bz = bias[ncol];
        #pragma unroll
        for (int r = 0; r < 8; ++r) {
            const int mr = m0 + r + hsel * 8;
            out[(size_t)mr * DD + ncol] = acc[t][r] + bz;
        }
    }
}

extern "C" void kernel_launch(void* const* d_in, const int* in_sizes, int n_in,
                              void* d_out, int out_size, void* d_ws, size_t ws_size,
                              hipStream_t stream) {
    const float* x    = (const float*)d_in[0];   // (B, N, D) fp32
    const float* sw   = (const float*)d_in[1];   // (12, N/2, 2, 2) fp32
    const float* Wout = (const float*)d_in[2];   // (D, D) fp32
    const float* bout = (const float*)d_in[3];   // (D,) fp32
    float* out = (float*)d_out;                  // (B, N, D) fp32

    // ws layout (ushort units): Yh | Yl | Wh | Wl  => ~34.6 MB total
    unsigned short* yh = (unsigned short*)d_ws;
    unsigned short* yl = yh + (size_t)BB * NN * DD;
    unsigned short* wh = yl + (size_t)BB * NN * DD;
    unsigned short* wl = wh + (size_t)DD * DD;

    // Split W_out into bf16 hi/lo planes (independent; tiny).
    wsplit_kernel<<<(DD * DD) / 256, 256, 0, stream>>>(Wout, wh, wl);

    dim3 bgrid(NN / 64, DD / 64, BB);            // (64, 8, 4)
    // Stages 0..5 : x -> out (f32 temp; overwritten later by GEMM)
    butterfly_p1_kernel<<<bgrid, 256, 0, stream>>>(x, out, sw);
    // Stages 6..11: out -> (Yh, Yl) bf16 planes; conversion paid once here
    butterfly_p2_kernel<<<bgrid, 256, 0, stream>>>(out, yh, yl, sw);

    // WMMA GEMM: (Yh,Yl,Wh,Wl) -> out
    dim3 ggrid((BB * NN) / 128, DD / 64);        // (128, 8)
    gemm_kernel<<<ggrid, 256, 0, stream>>>(yh, yl, wh, wl, bout, out);
}
